// Qwen3VLVisionAttentionPatch_12360915878785
// MI455X (gfx1250) — matrix operand order repaired and checked by reading, not compiled, not run on hardware
//
#include <hip/hip_runtime.h>

#define TOTAL 3072
#define HID   1280
#define HEADS 16
#define HDIM  80
#define DPAD  96
#define N3    3840
#define BSTR  36  // LDS B-tile row stride (floats): 16B-aligned, conflict-free

typedef __attribute__((ext_vector_type(2)))  float    v2f;
typedef __attribute__((ext_vector_type(4)))  float    v4f;
typedef __attribute__((ext_vector_type(8)))  float    v8f;
typedef __attribute__((ext_vector_type(16))) _Float16 v16h;
typedef __attribute__((ext_vector_type(8)))  _Float16 v8h;
typedef int v4i_vs __attribute__((vector_size(16)));  // matches builtin's V4i

union V16U { v16h v; v8h p[2]; };

#if defined(__has_builtin)
# if __has_builtin(__builtin_amdgcn_global_load_async_to_lds_b128)
#  define USE_ASYNC 1
# endif
#endif
#ifndef USE_ASYNC
# define USE_ASYNC 0
#endif

#if USE_ASYNC
#define ASYNC_CP_B128(gp, lp)                                                  \
  __builtin_amdgcn_global_load_async_to_lds_b128(                              \
      (__attribute__((address_space(1))) v4i_vs*)(gp),                         \
      (__attribute__((address_space(3))) v4i_vs*)(lp), 0, 0)
#endif

__device__ __forceinline__ v8f wmma_f32x4(v2f a, v2f b, v8f c) {
  // D = A(16x4,f32) * B(4x16,f32) + C(16x16,f32)
  return __builtin_amdgcn_wmma_f32_16x16x4_f32(false, a, false, b, (short)0, c,
                                               false, false);
}
__device__ __forceinline__ v8f wmma_f16(v16h a, v16h b, v8f c) {
  // D = A(16x32,f16) * B(32x16,f16) + C(16x16,f32)
  return __builtin_amdgcn_wmma_f32_16x16x32_f16(false, a, false, b, (short)0, c,
                                                false, false);
}

// ---------------------------------------------------------------------------
// f32 GEMM: out[M=3072, N] = A[M,1280] @ W[N,1280]^T + bias,  N in {3840,1280}
// Block = 128 threads = 4 waves -> 128x128 C macro-tile.
// All 4 waves share one 128-col B strip, staged in double-buffered LDS via
// async global->LDS copies (ASYNCcnt) when available. Per wave: 2 M-tiles x
// 8 N-tiles; per K-step(4): 2 global b64 (A) + 8 LDS b64 (B) + 16 WMMA.
// ---------------------------------------------------------------------------
__global__ void __launch_bounds__(128)
gemm_f32_lds(const float* __restrict__ A, const float* __restrict__ W,
             const float* __restrict__ bias, float* __restrict__ out,
             const int N) {
  __shared__ float bsm[2][128 * BSTR];  // 2 x 18.4KB
  const int NG   = N >> 7;              // N/128 n-groups
  const int ng   = blockIdx.x % NG;
  const int mg   = blockIdx.x / NG;
  const int tid  = threadIdx.x;
  const int lane = tid & 31;
  const int wid  = tid >> 5;
  const int lo = lane & 15, hi = lane >> 4;
  const int n0  = ng * 128;
  const int mt0 = (mg * 8 + wid * 2) * 16;  // this wave's first M row
  const float* arow0 = A + (size_t)(mt0 + lo) * HID;
  const float* arow1 = arow0 + (size_t)16 * HID;
  const float* wrow  = W + (size_t)(n0 + tid) * HID;  // row this thread copies

  v8f C0[8] = {}, C1[8] = {};
  const int NC = HID / 32;  // 40 K-chunks

#if USE_ASYNC
  // prologue: stream chunk 0 into buffer 0
#pragma unroll
  for (int i = 0; i < 8; ++i)
    ASYNC_CP_B128(wrow + i * 4, &bsm[0][tid * BSTR + i * 4]);
#else
  v4f stage[8];
#pragma unroll
  for (int i = 0; i < 8; ++i) stage[i] = *(const v4f*)(wrow + i * 4);
#endif

  for (int c = 0; c < NC; ++c) {
    const int buf = c & 1;
#if USE_ASYNC
    asm volatile("s_wait_asynccnt 0x0" ::: "memory");
    __syncthreads();  // chunk c fully in bsm[buf] for all waves
    if (c + 1 < NC) {
#pragma unroll
      for (int i = 0; i < 8; ++i)
        ASYNC_CP_B128(wrow + (c + 1) * 32 + i * 4,
                      &bsm[buf ^ 1][tid * BSTR + i * 4]);
    }
#else
    {
      float* bdst = &bsm[buf][tid * BSTR];
#pragma unroll
      for (int i = 0; i < 8; ++i) *(v4f*)(bdst + i * 4) = stage[i];
      if (c + 1 < NC) {
#pragma unroll
        for (int i = 0; i < 8; ++i)
          stage[i] = *(const v4f*)(wrow + (c + 1) * 32 + i * 4);
      }
    }
    __syncthreads();
#endif
    const float* bb = &bsm[buf][0];
#pragma unroll
    for (int ks = 0; ks < 8; ++ks) {
      const int kk = c * 32 + ks * 4;
      v2f a0 = *(const v2f*)(arow0 + kk + 2 * hi);
      v2f a1 = *(const v2f*)(arow1 + kk + 2 * hi);
#pragma unroll
      for (int j = 0; j < 8; ++j) {
        v2f b = *(const v2f*)(bb + (j * 16 + lo) * BSTR + ks * 4 + 2 * hi);
        C0[j] = wmma_f32x4(a0, b, C0[j]);
        C1[j] = wmma_f32x4(a1, b, C1[j]);
      }
    }
  }
#pragma unroll
  for (int j = 0; j < 8; ++j) {
    const int n = n0 + j * 16 + lo;
    const float bv = bias[n];
#pragma unroll
    for (int r = 0; r < 8; ++r) {
      out[(size_t)(mt0 + r + 8 * hi) * N + n]      = C0[j][r] + bv;
      out[(size_t)(mt0 + 16 + r + 8 * hi) * N + n] = C1[j][r] + bv;
    }
  }
}

// ---------------------------------------------------------------------------
// RoPE(q,k) (+ fold softmax scale into q), cast f16, relayout:
//   qp,kp: [h][t][96] (d 80..95 zero-padded), vT: [h][80][t]
// ---------------------------------------------------------------------------
__global__ void rope_pack(const float* __restrict__ qkv,
                          const float* __restrict__ cosb,
                          const float* __restrict__ sinb,
                          _Float16* __restrict__ qp, _Float16* __restrict__ kp,
                          _Float16* __restrict__ vT) {
  const int t = blockIdx.x;
  const float qscale = 0.11180339887498949f;  // 1/sqrt(80)
  const float* row = qkv + (size_t)t * N3;
  for (int i = threadIdx.x; i < HEADS * DPAD; i += blockDim.x) {
    int h = i / DPAD, d = i % DPAD;
    _Float16 qv = (_Float16)0.f, kv = (_Float16)0.f;
    if (d < HDIM) {
      float c = cosb[t * HDIM + d], s = sinb[t * HDIM + d];
      float qx = row[h * HDIM + d];
      float qr = (d < HDIM / 2) ? -row[h * HDIM + d + HDIM / 2]
                                :  row[h * HDIM + d - HDIM / 2];
      qv = (_Float16)((qx * c + qr * s) * qscale);
      float kx = row[HID + h * HDIM + d];
      float kr = (d < HDIM / 2) ? -row[HID + h * HDIM + d + HDIM / 2]
                                :  row[HID + h * HDIM + d - HDIM / 2];
      kv = (_Float16)(kx * c + kr * s);
    }
    qp[((size_t)h * TOTAL + t) * DPAD + d] = qv;
    kp[((size_t)h * TOTAL + t) * DPAD + d] = kv;
  }
  for (int i = threadIdx.x; i < HEADS * HDIM; i += blockDim.x) {
    int h = i / HDIM, d = i % HDIM;
    vT[((size_t)h * HDIM + d) * TOTAL + t] =
        (_Float16)row[2 * HID + h * HDIM + d];
  }
}

// ---------------------------------------------------------------------------
// Segment-masked flash attention, one wave per (head, 16-row qtile)
// ---------------------------------------------------------------------------
__global__ void __launch_bounds__(128)
flash_attn(const _Float16* __restrict__ qp, const _Float16* __restrict__ kp,
           const _Float16* __restrict__ vT, const int* __restrict__ cu,
           float* __restrict__ attn) {
  __shared__ _Float16 plds[4][16][32];
  const int lane = threadIdx.x & 31;
  const int wid  = threadIdx.x >> 5;
  const int task = blockIdx.x * 4 + wid;
  const int h    = task & (HEADS - 1);
  const int t0   = (task >> 4) * 16;
  const int lo = lane & 15, hi = lane >> 4;

  int cuv[7];
#pragma unroll
  for (int i = 0; i < 7; ++i) cuv[i] = cu[i];

  int rlo[8], rhi[8];
#pragma unroll
  for (int r = 0; r < 8; ++r) {
    int trow = t0 + r + 8 * hi;
    int a = 0, b = TOTAL;
#pragma unroll
    for (int s = 0; s < 6; ++s)
      if (trow >= cuv[s] && trow < cuv[s + 1]) { a = cuv[s]; b = cuv[s + 1]; }
    rlo[r] = a; rhi[r] = b;
  }
  int smin = 0, smax = TOTAL;
#pragma unroll
  for (int s = 0; s < 6; ++s) {
    if (t0 >= cuv[s] && t0 < cuv[s + 1]) smin = cuv[s];
    if (t0 + 15 >= cuv[s] && t0 + 15 < cuv[s + 1]) smax = cuv[s + 1];
  }

  // Q A-fragments: m=lo, K contiguous per lane half (16*hi..16*hi+15)
  V16U aq[3];
  const _Float16* qbase = qp + ((size_t)h * TOTAL + t0 + lo) * DPAD;
#pragma unroll
  for (int c = 0; c < 3; ++c) {
    aq[c].p[0] = *(const v8h*)(qbase + 32 * c + 16 * hi);
    aq[c].p[1] = *(const v8h*)(qbase + 32 * c + 16 * hi + 8);
  }

  float m8[8], l8[8];
  v8f O[5] = {};
#pragma unroll
  for (int r = 0; r < 8; ++r) { m8[r] = -3e38f; l8[r] = 0.f; }

  for (int s0 = (smin / 32) * 32; s0 < smax; s0 += 32) {
    v8f Sa = {}, Sb = {};
#pragma unroll
    for (int c = 0; c < 3; ++c) {
      // B: n=lo (s position), K rows 16*hi..+15 -> contiguous 32B in kp
      v16h bka = *(const v16h*)(kp + ((size_t)h * TOTAL + s0 + lo) * DPAD +
                                32 * c + 16 * hi);
      v16h bkb = *(const v16h*)(kp + ((size_t)h * TOTAL + s0 + 16 + lo) * DPAD +
                                32 * c + 16 * hi);
      Sa = wmma_f16(aq[c].v, bka, Sa);
      Sb = wmma_f16(aq[c].v, bkb, Sb);
    }
    const int sa_col = s0 + lo, sb_col = s0 + 16 + lo;
    float pa[8], pb[8], corr[8];
#pragma unroll
    for (int r = 0; r < 8; ++r) {
      float va = (sa_col >= rlo[r] && sa_col < rhi[r]) ? Sa[r] : -3e38f;
      float vb = (sb_col >= rlo[r] && sb_col < rhi[r]) ? Sb[r] : -3e38f;
      float mx = fmaxf(va, vb);
      mx = fmaxf(mx, __shfl_xor(mx, 1, 32));
      mx = fmaxf(mx, __shfl_xor(mx, 2, 32));
      mx = fmaxf(mx, __shfl_xor(mx, 4, 32));
      mx = fmaxf(mx, __shfl_xor(mx, 8, 32));
      float mn = fmaxf(m8[r], mx);
      float cr = __expf(m8[r] - mn);
      float ea = __expf(va - mn);
      float eb = __expf(vb - mn);
      float rs = ea + eb;
      rs += __shfl_xor(rs, 1, 32);
      rs += __shfl_xor(rs, 2, 32);
      rs += __shfl_xor(rs, 4, 32);
      rs += __shfl_xor(rs, 8, 32);
      l8[r] = l8[r] * cr + rs;
      m8[r] = mn;
      corr[r] = cr; pa[r] = ea; pb[r] = eb;
    }
#pragma unroll
    for (int d = 0; d < 5; ++d)
#pragma unroll
      for (int r = 0; r < 8; ++r) O[d][r] *= corr[r];

    // transpose P through LDS: C-layout (row-in-VGPR) -> A-layout (row-in-lane)
#pragma unroll
    for (int r = 0; r < 8; ++r) {
      plds[wid][r + 8 * hi][lo]      = (_Float16)pa[r];
      plds[wid][r + 8 * hi][16 + lo] = (_Float16)pb[r];
    }
    asm volatile("s_wait_dscnt 0x0" ::: "memory");
    V16U ap;
    ap.p[0] = *(const v8h*)&plds[wid][lo][16 * hi];
    ap.p[1] = *(const v8h*)&plds[wid][lo][16 * hi + 8];
#pragma unroll
    for (int d = 0; d < 5; ++d) {
      // B = V: n = d-col (dsub*16+lo), K rows = s0+16*hi..+15 contiguous in vT
      v16h bv = *(const v16h*)(vT + ((size_t)h * HDIM + d * 16 + lo) * TOTAL +
                               s0 + 16 * hi);
      O[d] = wmma_f16(ap.v, bv, O[d]);
    }
    asm volatile("s_wait_dscnt 0x0" ::: "memory");
  }
#pragma unroll
  for (int r = 0; r < 8; ++r) {
    float inv = 1.0f / l8[r];
    int trow = t0 + r + 8 * hi;
#pragma unroll
    for (int d = 0; d < 5; ++d)
      attn[(size_t)trow * HID + h * HDIM + d * 16 + lo] = O[d][r] * inv;
  }
}

// ---------------------------------------------------------------------------
extern "C" void kernel_launch(void* const* d_in, const int* in_sizes, int n_in,
                              void* d_out, int out_size, void* d_ws,
                              size_t ws_size, hipStream_t stream) {
  const float* hs     = (const float*)d_in[0];
  const float* qkv_w  = (const float*)d_in[1];
  const float* qkv_b  = (const float*)d_in[2];
  const float* proj_w = (const float*)d_in[3];
  const float* proj_b = (const float*)d_in[4];
  const float* cosb   = (const float*)d_in[5];
  const float* sinb   = (const float*)d_in[6];
  const int*   cu     = (const int*)d_in[7];
  (void)in_sizes; (void)n_in; (void)out_size; (void)ws_size;

  char* ws = (char*)d_ws;
  float* qkv = (float*)ws;  // 3072*3840*4 = 47.2MB (reused as attn output)
  size_t off = (size_t)TOTAL * N3 * sizeof(float);
  _Float16* qp = (_Float16*)(ws + off);
  off += (size_t)HEADS * TOTAL * DPAD * sizeof(_Float16);  // 9.44MB
  _Float16* kp = (_Float16*)(ws + off);
  off += (size_t)HEADS * TOTAL * DPAD * sizeof(_Float16);  // 9.44MB
  _Float16* vT = (_Float16*)(ws + off);                    // 7.86MB
  float* attn = qkv;  // qkv buffer dead after rope_pack

  // M groups: 192 mt-tiles / 8 per block = 24
  gemm_f32_lds<<<24 * (N3 / 128), 128, 0, stream>>>(hs, qkv_w, qkv_b, qkv, N3);
  rope_pack<<<TOTAL, 256, 0, stream>>>(qkv, cosb, sinb, qp, kp, vT);
  flash_attn<<<HEADS * (TOTAL / 16) / 4, 128, 0, stream>>>(qp, kp, vT, cu,
                                                           attn);
  gemm_f32_lds<<<24 * (HID / 128), 128, 0, stream>>>(attn, proj_w, proj_b,
                                                     (float*)d_out, HID);
}
